// TopkCompetitiveAttention_6768868458555
// MI455X (gfx1250) — matrix operand order repaired and checked by reading, not compiled, not run on hardware
//
#include <hip/hip_runtime.h>
#include <hip/hip_bf16.h>
#include <math.h>

#define B_ 16
#define Q_ 256
#define K_ 4096
#define DIM_ 1024
#define H_ 16
#define HD_ 64
#define QP_ 65     // kept queries (1 + top-64)
#define QPAD_ 80   // padded to multiple of 16 for WMMA tiles

// LDS-tiled GEMM geometry
#define BM_ 128    // block tile rows
#define BN_ 64     // block tile cols
#define KC_ 64     // k-chunk depth
#define LDS_STRIDE 72  // padded row stride (elements) to spread LDS banks; 144B, 16B-aligned

typedef __attribute__((ext_vector_type(16))) __bf16 v16bf;
typedef __attribute__((ext_vector_type(8)))  __bf16 v8bf;
typedef __attribute__((ext_vector_type(8)))  float  v8f;

// ---- float -> bf16 (round to nearest even) ----
__device__ __forceinline__ __bf16 f2bf(float f) {
  unsigned u = __builtin_bit_cast(unsigned, f);
  unsigned r = u + 0x7FFFu + ((u >> 16) & 1u);
  unsigned short h = (unsigned short)(r >> 16);
  return __builtin_bit_cast(__bf16, h);
}

// ---- async DMA: 16B per lane, global -> LDS (ASYNCcnt) ----
__device__ __forceinline__ void async_ld16(unsigned lds_off, const void* gptr) {
  unsigned long long ga = (unsigned long long)gptr;
  asm volatile("global_load_async_to_lds_b128 %0, %1, off"
               :: "v"(lds_off), "v"(ga) : "memory");
}
__device__ __forceinline__ void wait_async0() {
  asm volatile("s_wait_asynccnt 0x0" ::: "memory");
}

// ---- WMMA fragment loaders (ISA 7.12.2 16-bit layouts, wave32) ----
// A: 16x32 (MxK). lane<16: row=lane, K in {0..7,16..23}; lane>=16: row=lane-16, K in {8..15,24..31}
template <typename PT>
__device__ __forceinline__ v16bf load_a_frag(PT A, int lda, int lane) {
  int row = lane & 15;
  int kb  = (lane & 16) ? 8 : 0;
  PT p = A + (size_t)row * lda + kb;
  union { v16bf v; v8bf h[2]; } u;
  u.h[0] = *(const v8bf*)(p);        // K = kb .. kb+7
  u.h[1] = *(const v8bf*)(p + 16);   // K = kb+16 .. kb+23
  return u.v;
}
// B: 32x16 (KxN), source stored row-major as [N, Kinner] with stride ldb.
template <typename PT>
__device__ __forceinline__ v16bf load_b_frag(PT Bm, int ldb, int lane) {
  int col = lane & 15;
  int kb  = (lane & 16) ? 8 : 0;
  PT p = Bm + (size_t)col * ldb + kb;
  union { v16bf v; v8bf h[2]; } u;
  u.h[0] = *(const v8bf*)(p);        // K = kb .. kb+7
  u.h[1] = *(const v8bf*)(p + 16);   // K = kb+16 .. kb+23
  return u.v;
}

// One 16x64 tile k-step: 1 A fragment reused across 4 B fragments -> 4 WMMAs
template <typename PT>
__device__ __forceinline__ void wmma_16x64_step(PT Ap, PT Bp, int lda, int ldb,
                                                int lane, v8f acc[4]) {
  v16bf a = load_a_frag(Ap, lda, lane);
#pragma unroll
  for (int j = 0; j < 4; ++j) {
    v16bf b = load_b_frag(Bp + (size_t)(16 * j) * ldb, ldb, lane);
    acc[j] = __builtin_amdgcn_wmma_f32_16x16x32_bf16(false, a, false, b,
                                                     (short)0, acc[j], false, false);
  }
}

// ---- conversion ----
__global__ void cvt_kernel(const float* __restrict__ in, __bf16* __restrict__ out, size_t n) {
  size_t i = (size_t)blockIdx.x * blockDim.x + threadIdx.x;
  if (i < n) out[i] = f2bf(in[i]);
}

// ---- LDS-tiled GEMM: C[M,N] = A[M,Kd] * W[N,Kd]^T ----
// Block: 256 threads (8 waves), tile 128x64, k-chunks of 64, double-buffered async->LDS.
// MODE 0: store bf16 row-major [M,N], scaled
// MODE 1: v-projection transposed store -> vhT[B,H,HD,K]
// MODE 2: final projection: f32 + bias into d_out, dropping padded rows
template <int MODE>
__global__ void proj_gemm_kernel(const __bf16* __restrict__ A, const __bf16* __restrict__ W,
                                 __bf16* __restrict__ outb, float* __restrict__ outf,
                                 const float* __restrict__ bias,
                                 int M, int N, int Kd, int lda, int ldb, float scale) {
  __shared__ __bf16 sA[2][BM_ * LDS_STRIDE];   // 2 x 18KB
  __shared__ __bf16 sB[2][BN_ * LDS_STRIDE];   // 2 x 9KB

  const int t    = threadIdx.x;
  const int lane = t & 31;
  const int wv   = t >> 5;
  const int bn   = blockIdx.x % (N / BN_);
  const int bm   = blockIdx.x / (N / BN_);
  const int m0   = bm * BM_;
  const int n0   = bn * BN_;

  const __bf16* Ab = A + (size_t)m0 * lda;
  const __bf16* Bb = W + (size_t)n0 * ldb;

  const int trow = t >> 3;        // 0..31
  const int tchk = t & 7;         // 16B chunk within a 64-element row

  // stage k-chunk `kc` into buffer `buf`
  auto stage = [&](int buf, int kc) {
    unsigned baseA = (unsigned)(size_t)&sA[buf][0];
    unsigned baseB = (unsigned)(size_t)&sB[buf][0];
#pragma unroll
    for (int i = 0; i < BM_ / 32; ++i) {       // 4 issues: 128 rows
      int r = i * 32 + trow;
      async_ld16(baseA + (unsigned)(r * LDS_STRIDE + tchk * 8) * 2,
                 Ab + (size_t)r * lda + kc + tchk * 8);
    }
#pragma unroll
    for (int i = 0; i < BN_ / 32; ++i) {       // 2 issues: 64 rows
      int r = i * 32 + trow;
      async_ld16(baseB + (unsigned)(r * LDS_STRIDE + tchk * 8) * 2,
                 Bb + (size_t)r * ldb + kc + tchk * 8);
    }
  };

  v8f acc[4] = {};
  const int NC = Kd / KC_;        // 16 chunks for Kd=1024

  stage(0, 0);
  wait_async0();
  __syncthreads();

  for (int c = 0; c < NC; ++c) {
    if (c + 1 < NC) stage((c + 1) & 1, (c + 1) * KC_);
    const __bf16* sAp = &sA[c & 1][(wv * 16) * LDS_STRIDE];
    const __bf16* sBp = &sB[c & 1][0];
    wmma_16x64_step(sAp,      sBp,      LDS_STRIDE, LDS_STRIDE, lane, acc);
    wmma_16x64_step(sAp + 32, sBp + 32, LDS_STRIDE, LDS_STRIDE, lane, acc);
    if (c + 1 < NC) { wait_async0(); __syncthreads(); }
  }

  const int col = lane & 15;
  const int mb  = (lane & 16) ? 8 : 0;
  const int mw0 = m0 + wv * 16;
#pragma unroll
  for (int j = 0; j < 4; ++j) {
#pragma unroll
    for (int i = 0; i < 8; ++i) {
      int m = mw0 + mb + i;
      int n = n0 + 16 * j + col;
      float vr = acc[j][i] * scale;
      if (MODE == 0) {
        outb[(size_t)m * N + n] = f2bf(vr);
      } else if (MODE == 1) {              // vhT[b][h][c][k]
        int bb = m >> 12;                  // m / K_
        int kr = m & (K_ - 1);
        int h = n >> 6, cc = n & 63;
        outb[((size_t)(bb * H_ + h) * HD_ + cc) * K_ + kr] = f2bf(vr);
      } else {                             // MODE 2
        int bb = m / QPAD_, qp = m % QPAD_;
        if (qp < QP_) outf[((size_t)(bb * QP_ + qp)) * DIM_ + n] = vr + bias[n];
      }
    }
  }
}

// ---- dots[b,h,q,k] = (qh/scale) . kh over head slice (inner=64) ----
__global__ void dots_kernel(const __bf16* __restrict__ qh, const __bf16* __restrict__ kh,
                            float* __restrict__ dots) {
  int lane = threadIdx.x & 31;
  int wg   = blockIdx.x * 8 + (threadIdx.x >> 5);
  const int ktiles = K_ / 64;   // 64
  const int qtiles = Q_ / 16;   // 16
  int per_bh = qtiles * ktiles; // 1024
  int bh = wg / per_bh;
  int r  = wg % per_bh;
  int qt = r / ktiles, kt = r % ktiles;
  int b = bh / H_, h = bh % H_;
  const __bf16* Ap = qh + (size_t)(b * Q_ + qt * 16) * DIM_ + h * HD_;
  const __bf16* Bp = kh + (size_t)(b * K_ + kt * 64) * DIM_ + h * HD_;
  v8f acc[4] = {};
  wmma_16x64_step(Ap,      Bp,      DIM_, DIM_, lane, acc);
  wmma_16x64_step(Ap + 32, Bp + 32, DIM_, DIM_, lane, acc);
  int col = lane & 15;
  int mb  = (lane & 16) ? 8 : 0;
#pragma unroll
  for (int j = 0; j < 4; ++j)
#pragma unroll
    for (int i = 0; i < 8; ++i)
      dots[((size_t)bh * Q_ + qt * 16 + mb + i) * K_ + kt * 64 + 16 * j + col] = acc[j][i];
}

// ---- off[b,h,k] = colmax + log(colsum) over q=1..Q-1 ----
__global__ void colstats_kernel(const float* __restrict__ dots, float* __restrict__ off) {
  size_t tid = (size_t)blockIdx.x * blockDim.x + threadIdx.x; // < B*H*K
  int bh = (int)(tid / K_);
  int k  = (int)(tid % K_);
  const float* p = dots + (size_t)bh * Q_ * K_ + k;
  float m = -1e30f, s = 0.f;
  for (int q = 1; q < Q_; ++q) {
    float d  = p[(size_t)q * K_];
    float nm = fmaxf(m, d);
    s = s * __expf(m - nm) + __expf(d - nm);
    m = nm;
  }
  off[tid] = m + __logf(s);
}

// ---- scoreL[b,q] = max_{h,k} (dots - off); monotone-equivalent to reference score ----
__global__ void score_kernel(const float* __restrict__ dots, const float* __restrict__ off,
                             float* __restrict__ scoreL) {
  __shared__ float red[256];
  int b = blockIdx.x / Q_, q = blockIdx.x % Q_;
  if (q == 0) { if (threadIdx.x == 0) scoreL[blockIdx.x] = -1e30f; return; }
  float m = -1e30f;
  for (int t = threadIdx.x; t < H_ * K_; t += 256) {
    int h = t >> 12;            // t / K_
    int k = t & (K_ - 1);
    size_t bh = (size_t)(b * H_ + h);
    float d = dots[(bh * Q_ + q) * K_ + k] - off[bh * K_ + k];
    m = fmaxf(m, d);
  }
  red[threadIdx.x] = m; __syncthreads();
  for (int s = 128; s > 0; s >>= 1) {
    if (threadIdx.x < s) red[threadIdx.x] = fmaxf(red[threadIdx.x], red[threadIdx.x + s]);
    __syncthreads();
  }
  if (threadIdx.x == 0) scoreL[blockIdx.x] = red[0];
}

// ---- per-batch top-64 (descending, tie -> lowest index), keep[0]=0 ----
__global__ void topk_kernel(const float* __restrict__ scoreL, int* __restrict__ keep,
                            int* __restrict__ keep_out) {
  __shared__ float val[256];
  __shared__ float rv[256];
  __shared__ int   ri[256];
  int b = blockIdx.x, t = threadIdx.x;
  val[t] = (t == 0) ? -1e30f : scoreL[b * Q_ + t];
  if (t == 0) { keep[b * QP_] = 0; keep_out[b * QP_] = 0; }
  __syncthreads();
  for (int it = 0; it < 64; ++it) {
    rv[t] = val[t]; ri[t] = t;
    __syncthreads();
    for (int s = 128; s > 0; s >>= 1) {
      if (t < s) {
        if (rv[t + s] > rv[t] || (rv[t + s] == rv[t] && ri[t + s] < ri[t])) {
          rv[t] = rv[t + s]; ri[t] = ri[t + s];
        }
      }
      __syncthreads();
    }
    if (t == 0) {
      int j = ri[0];
      keep[b * QP_ + 1 + it] = j;
      keep_out[b * QP_ + 1 + it] = j;
      val[j] = -1e30f;
    }
    __syncthreads();
  }
}

// ---- gather kept rows + softmax over the 65 kept queries; bf16, padded to 80 rows ----
__global__ void attnsm_kernel(const float* __restrict__ dots, const int* __restrict__ keep,
                              __bf16* __restrict__ attn) {
  __shared__ int kp[QP_];
  size_t tid = (size_t)blockIdx.x * 256 + threadIdx.x; // < B*H*K
  int bh = (int)(tid / K_);
  int k  = (int)(tid % K_);
  int b  = bh / H_;
  if (threadIdx.x < QP_) kp[threadIdx.x] = keep[b * QP_ + threadIdx.x];
  __syncthreads();
  const float* p = dots + (size_t)bh * Q_ * K_ + k;
  float m = -1e30f, s = 0.f;
  for (int q = 0; q < QP_; ++q) {
    float d  = p[(size_t)kp[q] * K_];
    float nm = fmaxf(m, d);
    s = s * __expf(m - nm) + __expf(d - nm);
    m = nm;
  }
  float inv = 1.f / s;
  __bf16* o = attn + (size_t)bh * QPAD_ * K_ + k;
  for (int q = 0; q < QP_; ++q) {
    float d = p[(size_t)kp[q] * K_];
    o[(size_t)q * K_] = f2bf(__expf(d - m) * inv);
  }
  for (int q = QP_; q < QPAD_; ++q) o[(size_t)q * K_] = f2bf(0.f);
}

// ---- out[b,q',h,c] = attn[b,h,q',:] . vhT[b,h,c,:]  (inner K=4096) ----
__global__ void attnv_kernel(const __bf16* __restrict__ attn, const __bf16* __restrict__ vhT,
                             __bf16* __restrict__ outc) {
  int lane = threadIdx.x & 31;
  int wg   = blockIdx.x * 8 + (threadIdx.x >> 5);
  int bh = wg / (QPAD_ / 16);
  int mt = wg % (QPAD_ / 16);
  int b = bh / H_, h = bh % H_;
  const __bf16* Ap = attn + ((size_t)bh * QPAD_ + mt * 16) * K_;
  const __bf16* Bp = vhT + (size_t)bh * HD_ * K_;
  v8f acc[4] = {};
  for (int k0 = 0; k0 < K_; k0 += 32)
    wmma_16x64_step(Ap + k0, Bp + k0, K_, K_, lane, acc);
  int col = lane & 15;
  int mb  = (lane & 16) ? 8 : 0;
#pragma unroll
  for (int j = 0; j < 4; ++j)
#pragma unroll
    for (int i = 0; i < 8; ++i) {
      int m = mt * 16 + mb + i;
      int n = 16 * j + col;
      outc[((size_t)(b * QPAD_) + m) * DIM_ + h * HD_ + n] = f2bf(acc[j][i]);
    }
}

extern "C" void kernel_launch(void* const* d_in, const int* in_sizes, int n_in,
                              void* d_out, int out_size, void* d_ws, size_t ws_size,
                              hipStream_t stream) {
  const float* q  = (const float*)d_in[0];
  const float* k  = (const float*)d_in[1];
  const float* v  = (const float*)d_in[2];
  const float* Wq = (const float*)d_in[3];
  const float* Wk = (const float*)d_in[4];
  const float* Wv = (const float*)d_in[5];
  const float* Wo = (const float*)d_in[6];
  const float* bo = (const float*)d_in[7];
  (void)in_sizes; (void)n_in; (void)out_size; (void)ws_size;

  const size_t nq = (size_t)B_ * Q_ * DIM_;     // 4,194,304
  const size_t nk = (size_t)B_ * K_ * DIM_;     // 67,108,864
  const size_t nw = (size_t)DIM_ * DIM_;        // 1,048,576

  char* ws = (char*)d_ws;
  size_t off_b = 0;
  auto alloc = [&](size_t bytes) -> void* {
    void* p = ws + off_b;
    off_b = (off_b + bytes + 255) & ~(size_t)255;
    return p;
  };
  __bf16* qb   = (__bf16*)alloc(nq * 2);
  __bf16* kb   = (__bf16*)alloc(nk * 2);
  __bf16* vb   = (__bf16*)alloc(nk * 2);
  __bf16* wqb  = (__bf16*)alloc(nw * 2);
  __bf16* wkb  = (__bf16*)alloc(nw * 2);
  __bf16* wvb  = (__bf16*)alloc(nw * 2);
  __bf16* wob  = (__bf16*)alloc(nw * 2);
  __bf16* qh   = (__bf16*)alloc(nq * 2);                         // scaled by 1/sqrt(HD)
  __bf16* kh   = (__bf16*)alloc(nk * 2);
  __bf16* vhT  = (__bf16*)alloc(nk * 2);                         // [B,H,HD,K]
  float*  dots = (float*)alloc((size_t)B_ * H_ * Q_ * K_ * 4);   // 1 GB
  float*  offs = (float*)alloc((size_t)B_ * H_ * K_ * 4);
  float*  scoreL = (float*)alloc((size_t)B_ * Q_ * 4);
  int*    keep = (int*)alloc((size_t)B_ * QP_ * 4);
  __bf16* attn = (__bf16*)alloc((size_t)B_ * H_ * QPAD_ * K_ * 2);
  __bf16* outc = (__bf16*)alloc((size_t)B_ * QPAD_ * DIM_ * 2);

  auto cvt = [&](const float* in, __bf16* outp, size_t n) {
    cvt_kernel<<<(unsigned)((n + 255) / 256), 256, 0, stream>>>(in, outp, n);
  };
  cvt(q, qb, nq); cvt(k, kb, nk); cvt(v, vb, nk);
  cvt(Wq, wqb, nw); cvt(Wk, wkb, nw); cvt(Wv, wvb, nw); cvt(Wo, wob, nw);

  const float INV_SCALE = 0.125f;  // 1/sqrt(64)

  { int M = B_ * Q_, blocks = (M / BM_) * (DIM_ / BN_);
    proj_gemm_kernel<0><<<blocks, 256, 0, stream>>>(qb, wqb, qh, nullptr, nullptr,
                                                    M, DIM_, DIM_, DIM_, DIM_, INV_SCALE); }
  { int M = B_ * K_, blocks = (M / BM_) * (DIM_ / BN_);
    proj_gemm_kernel<0><<<blocks, 256, 0, stream>>>(kb, wkb, kh, nullptr, nullptr,
                                                    M, DIM_, DIM_, DIM_, DIM_, 1.f); }
  { int M = B_ * K_, blocks = (M / BM_) * (DIM_ / BN_);
    proj_gemm_kernel<1><<<blocks, 256, 0, stream>>>(vb, wvb, vhT, nullptr, nullptr,
                                                    M, DIM_, DIM_, DIM_, DIM_, 1.f); }

  { int waves = B_ * H_ * (Q_ / 16) * (K_ / 64);
    dots_kernel<<<waves / 8, 256, 0, stream>>>(qh, kh, dots); }

  colstats_kernel<<<(B_ * H_ * K_) / 256, 256, 0, stream>>>(dots, offs);
  score_kernel<<<B_ * Q_, 256, 0, stream>>>(dots, offs, scoreL);

  float* out_f    = (float*)d_out;
  int*   keep_out = (int*)d_out + (size_t)B_ * QP_ * DIM_;
  topk_kernel<<<B_, 256, 0, stream>>>(scoreL, keep, keep_out);

  attnsm_kernel<<<(B_ * H_ * K_) / 256, 256, 0, stream>>>(dots, keep, attn);

  { int waves = B_ * H_ * (QPAD_ / 16);
    attnv_kernel<<<waves / 8, 256, 0, stream>>>(attn, vhT, outc); }

  { int M = B_ * QPAD_, blocks = (M / BM_) * (DIM_ / BN_);
    proj_gemm_kernel<2><<<blocks, 256, 0, stream>>>(outc, wob, nullptr, out_f, bo,
                                                    M, DIM_, DIM_, DIM_, DIM_, 1.f); }
}
